// Gemma3Attention_11123965296802
// MI455X (gfx1250) — matrix operand order repaired and checked by reading, not compiled, not run on hardware
//
#include <hip/hip_runtime.h>
#include <hip/hip_fp16.h>

typedef __attribute__((ext_vector_type(16))) _Float16 v16h;
typedef __attribute__((ext_vector_type(8)))  float    v8f;
typedef int b128v __attribute__((vector_size(16)));   // matches builtin param pointee

union Frag { v16h h; float4 f4[2]; };

#define S_LEN 2048
#define HIDW  2560
#define NH    8
#define NKV   4
#define HD    256
#define WIN   512
#define ATTN_SCALE 0.0625f   // 256^-0.5
#define SOFTCAP    50.0f

#if defined(__has_builtin)
#if __has_builtin(__builtin_amdgcn_global_load_async_to_lds_b128)
#define ASYNC_LDS 1
#endif
#endif

__device__ __forceinline__ v8f wmma_f16(const Frag& a, const Frag& b, v8f c) {
  return __builtin_amdgcn_wmma_f32_16x16x32_f16(false, a.h, false, b.h,
                                                (short)0, c, false, false);
}

// 16-byte global -> LDS copy: async (ASYNCcnt-tracked, VGPR-bypassing) on
// CDNA5 when the builtin is available; VGPR round-trip fallback otherwise.
__device__ __forceinline__ void g2l_16B(const _Float16* g, _Float16* l) {
#ifdef ASYNC_LDS
  __builtin_amdgcn_global_load_async_to_lds_b128(
      (__attribute__((address_space(1))) b128v*)(unsigned long long)g,
      (__attribute__((address_space(3))) b128v*)(unsigned)(unsigned long long)l,
      0, 0);
#else
  *(float4*)l = *(const float4*)g;
#endif
}

__device__ __forceinline__ void g2l_fence() {
#ifdef ASYNC_LDS
#if __has_builtin(__builtin_amdgcn_s_wait_asynccnt)
  __builtin_amdgcn_s_wait_asynccnt(0);
#else
  asm volatile("s_wait_asynccnt 0x0" ::: "memory");
#endif
#endif
}

// ---------------------------------------------------------------------------
// fp32 -> f16 conversion
// ---------------------------------------------------------------------------
__global__ void cvt_f16_kernel(const float* __restrict__ s,
                               _Float16* __restrict__ d, int n) {
  int i = blockIdx.x * 256 + threadIdx.x;
  if (i < n) d[i] = (_Float16)s[i];
}

// ---------------------------------------------------------------------------
// C[M,N] = A[M,K] * B[N,K]^T   (f16 in, fp32 out)
// block: 256 threads = 8 waves; tile 128(M) x 256(N); K-step 32 via LDS
// waves arranged 2(M) x 4(N); each wave owns 64x64 = 4x4 WMMA accums
// -> 16 v_wmma per 16 ds_load_b128 per K-step (1 LDS load per WMMA)
// ---------------------------------------------------------------------------
__global__ __launch_bounds__(256)
void gemm_nt_f16(const _Float16* __restrict__ A, const _Float16* __restrict__ B,
                 float* __restrict__ C, int M, int N, int K) {
  __shared__ _Float16 As[128 * 32];   //  8 KB
  __shared__ _Float16 Bs[256 * 32];   // 16 KB
  const int tid  = threadIdx.x;
  const int lane = tid & 31;
  const int wid  = tid >> 5;
  const int ln = lane & 15, hi = lane >> 4;
  const int wm = wid & 1, wn = wid >> 1;
  const int m0 = blockIdx.y * 128, n0 = blockIdx.x * 256;

  v8f acc[4][4] = {};

  for (int k0 = 0; k0 < K; k0 += 32) {
    // cooperative stage: A tile 128x32 (512 float4), B tile 256x32 (1024 float4)
#pragma unroll
    for (int t = 0; t < 2; t++) {
      const int i = tid + t * 256;
      const int row = i >> 2, vec = i & 3;
      g2l_16B(A + (size_t)(m0 + row) * K + k0 + vec * 8,
              &As[row * 32 + vec * 8]);
    }
#pragma unroll
    for (int t = 0; t < 4; t++) {
      const int i = tid + t * 256;
      const int row = i >> 2, vec = i & 3;
      g2l_16B(B + (size_t)(n0 + row) * K + k0 + vec * 8,
              &Bs[row * 32 + vec * 8]);
    }
    g2l_fence();
    __syncthreads();

    Frag a[4], b[4];
#pragma unroll
    for (int mi = 0; mi < 4; mi++) {
      const _Float16* p = &As[(wm * 64 + mi * 16 + ln) * 32];
      a[mi].f4[0] = *(const float4*)(p + hi * 8);        // K = hi*8 .. +7
      a[mi].f4[1] = *(const float4*)(p + 16 + hi * 8);   // K = 16+hi*8 .. +7
    }
#pragma unroll
    for (int ni = 0; ni < 4; ni++) {
      const _Float16* p = &Bs[(wn * 64 + ni * 16 + ln) * 32 + hi * 8];
      b[ni].f4[0] = *(const float4*)p;                   // K = hi*8 .. +7
      b[ni].f4[1] = *(const float4*)(p + 16);            // K = 16+hi*8 .. +7
    }
#pragma unroll
    for (int mi = 0; mi < 4; mi++)
#pragma unroll
      for (int ni = 0; ni < 4; ni++)
        acc[mi][ni] = wmma_f16(a[mi], b[ni], acc[mi][ni]);
    __syncthreads();
  }

#pragma unroll
  for (int mi = 0; mi < 4; mi++)
#pragma unroll
    for (int ni = 0; ni < 4; ni++)
#pragma unroll
      for (int r = 0; r < 8; r++) {
        const int row = m0 + wm * 64 + mi * 16 + r + 8 * hi;
        const int col = n0 + wn * 64 + ni * 16 + ln;
        C[(size_t)row * N + col] = acc[mi][ni][r];
      }
}

// ---------------------------------------------------------------------------
// RMSNorm + RoPE; emits f16 Q[H][S][D], K[KV][S][D], V^T[KV][D][S]
// one block (256 thr) per token s; thread d handles dim d of each head
// ---------------------------------------------------------------------------
__global__ __launch_bounds__(256)
void rmsrope_kernel(const float* __restrict__ qf, const float* __restrict__ kf,
                    const float* __restrict__ vf,
                    const float* __restrict__ cosT, const float* __restrict__ sinT,
                    const float* __restrict__ qw, const float* __restrict__ kw,
                    _Float16* __restrict__ Qh, _Float16* __restrict__ Kh,
                    _Float16* __restrict__ Vt) {
  const int s = blockIdx.x;
  const int d = threadIdx.x;
  __shared__ float sh[256];
  const float c  = cosT[(size_t)s * HD + d];
  const float sn = sinT[(size_t)s * HD + d];
  const float qwd = 1.f + qw[d];
  const float kwd = 1.f + kw[d];

  for (int h = 0; h < NH; h++) {
    float x = qf[(size_t)s * (NH * HD) + h * HD + d];
    sh[d] = x * x; __syncthreads();
    for (int o = 128; o > 0; o >>= 1) { if (d < o) sh[d] += sh[d + o]; __syncthreads(); }
    const float inv = rsqrtf(sh[0] / (float)HD + 1e-6f);
    __syncthreads();
    const float xn = x * inv * qwd;
    sh[d] = xn; __syncthreads();
    const float rot = (d < 128) ? -sh[d + 128] : sh[d - 128];
    Qh[((size_t)h * S_LEN + s) * HD + d] = (_Float16)(xn * c + rot * sn);
    __syncthreads();
  }
  for (int g = 0; g < NKV; g++) {
    float x = kf[(size_t)s * (NKV * HD) + g * HD + d];
    sh[d] = x * x; __syncthreads();
    for (int o = 128; o > 0; o >>= 1) { if (d < o) sh[d] += sh[d + o]; __syncthreads(); }
    const float inv = rsqrtf(sh[0] / (float)HD + 1e-6f);
    __syncthreads();
    const float xn = x * inv * kwd;
    sh[d] = xn; __syncthreads();
    const float rot = (d < 128) ? -sh[d + 128] : sh[d - 128];
    Kh[((size_t)g * S_LEN + s) * HD + d] = (_Float16)(xn * c + rot * sn);
    __syncthreads();
    Vt[((size_t)g * HD + d) * S_LEN + s] =
        (_Float16)vf[(size_t)s * (NKV * HD) + g * HD + d];
  }
}

// ---------------------------------------------------------------------------
// Windowed causal flash attention with softcap. One wave per (head, 16 rows).
// Oh = f16 [S][H*D]
// ---------------------------------------------------------------------------
__global__ __launch_bounds__(32)
void attn_kernel(const _Float16* __restrict__ Qh, const _Float16* __restrict__ Kh,
                 const _Float16* __restrict__ Vt, _Float16* __restrict__ Oh) {
  const int qt = blockIdx.x, h = blockIdx.y;
  const int q0 = qt * 16;
  const int g  = h >> 1;                 // H/KV = 2
  const int lane = threadIdx.x, ln = lane & 15, hi = lane >> 4;
  __shared__ _Float16 Pl[16 * 32];

  // Q fragments for rows q0..q0+15, all 8 K-steps of D=256, kept in VGPRs
  Frag qa[8];
  const _Float16* Qbase = Qh + ((size_t)h * S_LEN + q0 + ln) * HD;
#pragma unroll
  for (int kb = 0; kb < 8; kb++) {
    qa[kb].f4[0] = *(const float4*)(Qbase + kb * 32 + hi * 8);
    qa[kb].f4[1] = *(const float4*)(Qbase + kb * 32 + 16 + hi * 8);
  }

  v8f oacc[16] = {};
  float mrow[8], lrow[8];
#pragma unroll
  for (int r = 0; r < 8; r++) { mrow[r] = -1e30f; lrow[r] = 0.f; }

  const int jend = q0 + 15;
  int js = q0 - (WIN - 1); if (js < 0) js = 0; js &= ~31;

  for (int j0 = js; j0 <= jend; j0 += 32) {
    // ---- scores: S = Q K^T for keys j0..j0+31 (two 16x16 tiles) ----
    v8f s1 = {}, s2 = {};
    int jr1 = j0 + ln;      if (jr1 > S_LEN - 1) jr1 = S_LEN - 1;
    int jr2 = j0 + 16 + ln; if (jr2 > S_LEN - 1) jr2 = S_LEN - 1;
    const _Float16* K1 = Kh + ((size_t)g * S_LEN + jr1) * HD + hi * 8;
    const _Float16* K2 = Kh + ((size_t)g * S_LEN + jr2) * HD + hi * 8;
#pragma unroll
    for (int kb = 0; kb < 8; kb++) {
      Frag b1, b2;
      b1.f4[0] = *(const float4*)(K1 + kb * 32);
      b1.f4[1] = *(const float4*)(K1 + kb * 32 + 16);
      b2.f4[0] = *(const float4*)(K2 + kb * 32);
      b2.f4[1] = *(const float4*)(K2 + kb * 32 + 16);
      s1 = wmma_f16(qa[kb], b1, s1);
      s2 = wmma_f16(qa[kb], b2, s2);
    }

    // ---- softcap + mask + online softmax (row M = r + 8*hi, col N = ln) ----
    float scalef[8], p1v[8], p2v[8];
#pragma unroll
    for (int r = 0; r < 8; r++) {
      const int qrow = q0 + r + 8 * hi;
      const int j1 = j0 + ln, j2 = j0 + 16 + ln;
      const bool v1 = (j1 <= qrow) && (qrow - j1 < WIN);
      const bool v2 = (j2 <= qrow) && (qrow - j2 < WIN);
      float x1 = SOFTCAP * tanhf(s1[r] * (ATTN_SCALE / SOFTCAP));
      float x2 = SOFTCAP * tanhf(s2[r] * (ATTN_SCALE / SOFTCAP));
      x1 = v1 ? x1 : -1e30f;
      x2 = v2 ? x2 : -1e30f;
      float t = fmaxf(x1, x2);
      t = fmaxf(t, __shfl_xor(t, 1));
      t = fmaxf(t, __shfl_xor(t, 2));
      t = fmaxf(t, __shfl_xor(t, 4));
      t = fmaxf(t, __shfl_xor(t, 8));
      const float mnew = fmaxf(mrow[r], t);
      const float p1 = v1 ? __expf(x1 - mnew) : 0.f;
      const float p2 = v2 ? __expf(x2 - mnew) : 0.f;
      float su = p1 + p2;
      su += __shfl_xor(su, 1);
      su += __shfl_xor(su, 2);
      su += __shfl_xor(su, 4);
      su += __shfl_xor(su, 8);
      scalef[r] = __expf(mrow[r] - mnew);
      lrow[r] = lrow[r] * scalef[r] + su;
      mrow[r] = mnew;
      p1v[r] = p1; p2v[r] = p2;
    }
#pragma unroll
    for (int nt = 0; nt < 16; nt++)
#pragma unroll
      for (int r = 0; r < 8; r++)
        oacc[nt][r] *= scalef[r];

    // ---- transpose P (C-layout -> A-layout) through LDS ----
#pragma unroll
    for (int r = 0; r < 8; r++) {
      Pl[(r + 8 * hi) * 32 + ln]      = (_Float16)p1v[r];
      Pl[(r + 8 * hi) * 32 + 16 + ln] = (_Float16)p2v[r];
    }
    __syncthreads();
    Frag pa;
    pa.f4[0] = *(const float4*)&Pl[ln * 32 + hi * 8];
    pa.f4[1] = *(const float4*)&Pl[ln * 32 + 16 + hi * 8];
    __syncthreads();

    // ---- O += P * V  (B from V^T[KV][D][S]: contiguous over key) ----
    const _Float16* Vbase = Vt + ((size_t)g * HD + ln) * S_LEN + j0 + hi * 8;
#pragma unroll
    for (int nt = 0; nt < 16; nt++) {
      Frag bv;
      const _Float16* p = Vbase + (size_t)(nt * 16) * S_LEN;
      bv.f4[0] = *(const float4*)p;
      bv.f4[1] = *(const float4*)(p + 16);
      oacc[nt] = wmma_f16(pa, bv, oacc[nt]);
    }
  }

#pragma unroll
  for (int nt = 0; nt < 16; nt++)
#pragma unroll
    for (int r = 0; r < 8; r++) {
      const int row = q0 + r + 8 * hi;
      const int col = h * HD + nt * 16 + ln;
      Oh[(size_t)row * (NH * HD) + col] = (_Float16)(oacc[nt][r] / lrow[r]);
    }
}

// ---------------------------------------------------------------------------
extern "C" void kernel_launch(void* const* d_in, const int* in_sizes, int n_in,
                              void* d_out, int out_size, void* d_ws, size_t ws_size,
                              hipStream_t stream) {
  (void)in_sizes; (void)n_in; (void)out_size; (void)ws_size;
  const float* X    = (const float*)d_in[0];
  // d_in[1] = position_ids (arange, used implicitly)
  const float* cosT = (const float*)d_in[2];
  const float* sinT = (const float*)d_in[3];
  const float* Wq   = (const float*)d_in[4];
  const float* Wk   = (const float*)d_in[5];
  const float* Wv   = (const float*)d_in[6];
  const float* Wo   = (const float*)d_in[7];
  const float* qw   = (const float*)d_in[8];
  const float* kw   = (const float*)d_in[9];
  float* out = (float*)d_out;

  char* ws = (char*)d_ws;
  size_t off = 0;
  auto alloc = [&](size_t bytes) {
    void* p = ws + off;
    off += (bytes + 255) & ~(size_t)255;
    return p;
  };
  _Float16* Xh  = (_Float16*)alloc((size_t)S_LEN * HIDW * 2);
  _Float16* Wqh = (_Float16*)alloc((size_t)NH  * HD * HIDW * 2);
  _Float16* Wkh = (_Float16*)alloc((size_t)NKV * HD * HIDW * 2);
  _Float16* Wvh = (_Float16*)alloc((size_t)NKV * HD * HIDW * 2);
  _Float16* Woh = (_Float16*)alloc((size_t)HIDW * (NH * HD) * 2);
  float*    qf  = (float*)  alloc((size_t)S_LEN * (NH  * HD) * 4);
  float*    kf  = (float*)  alloc((size_t)S_LEN * (NKV * HD) * 4);
  float*    vf  = (float*)  alloc((size_t)S_LEN * (NKV * HD) * 4);
  _Float16* Qh  = (_Float16*)alloc((size_t)NH  * S_LEN * HD * 2);
  _Float16* Kh  = (_Float16*)alloc((size_t)NKV * S_LEN * HD * 2);
  _Float16* Vt  = (_Float16*)alloc((size_t)NKV * HD * S_LEN * 2 + 256); // +pad for tail reads
  _Float16* Oh  = (_Float16*)alloc((size_t)S_LEN * (NH * HD) * 2);

  auto cvt = [&](const float* src, _Float16* dst, int n) {
    cvt_f16_kernel<<<(n + 255) / 256, 256, 0, stream>>>(src, dst, n);
  };
  cvt(X,  Xh,  S_LEN * HIDW);
  cvt(Wq, Wqh, NH  * HD * HIDW);
  cvt(Wk, Wkh, NKV * HD * HIDW);
  cvt(Wv, Wvh, NKV * HD * HIDW);
  cvt(Wo, Woh, HIDW * NH * HD);

  // QKV projections
  gemm_nt_f16<<<dim3((NH  * HD) / 256, S_LEN / 128), 256, 0, stream>>>(
      Xh, Wqh, qf, S_LEN, NH * HD, HIDW);
  gemm_nt_f16<<<dim3((NKV * HD) / 256, S_LEN / 128), 256, 0, stream>>>(
      Xh, Wkh, kf, S_LEN, NKV * HD, HIDW);
  gemm_nt_f16<<<dim3((NKV * HD) / 256, S_LEN / 128), 256, 0, stream>>>(
      Xh, Wvh, vf, S_LEN, NKV * HD, HIDW);

  // RMSNorm + RoPE + V transpose
  rmsrope_kernel<<<S_LEN, 256, 0, stream>>>(qf, kf, vf, cosT, sinT, qw, kw,
                                            Qh, Kh, Vt);

  // Windowed causal attention
  attn_kernel<<<dim3(S_LEN / 16, NH), 32, 0, stream>>>(Qh, Kh, Vt, Oh);

  // Output projection
  gemm_nt_f16<<<dim3(HIDW / 256, S_LEN / 128), 256, 0, stream>>>(
      Oh, Woh, out, S_LEN, HIDW, NH * HD);
}
